// MultiHeadAttention_35570919146065
// MI455X (gfx1250) — compile-verified, matrix operand order checked by reading, not run on hardware
//
#include <hip/hip_runtime.h>
#include <hip/hip_bf16.h>

// ---------------------------------------------------------------------------
// MHA with relative position bias on gfx1250 (MI455X), bf16 WMMA path.
//   B=2, S=2048, D=1024, H=16, DK=64
// Pipeline:
//   1) fp32 -> bf16 convert of q,k,v activations
//   2) fp32 -> bf16 transpose-convert of Wq,Wk,Wv,Wo  (N-major for WMMA B operand)
//   3) rel_sum[d] = sum_64 rel_table[d][:]            (bias is a function of j-i only)
//   4) WMMA GEMM: Q/K/V projections (bf16 out, +bias) — async-to-LDS double-buffered
//   5) flash-attention kernel (WMMA for QK^T and PV, online softmax, causal+rel bias)
//   6) WMMA GEMM: output projection (f32 out, +bias) -> d_out
// ---------------------------------------------------------------------------

#define B_  2
#define S_  2048
#define D_  1024
#define H_  16
#define DK_ 64
#define MAXREL_ 2047

typedef __attribute__((ext_vector_type(16))) __bf16 v16bf;
typedef __attribute__((ext_vector_type(8)))  float  v8f;

union Frag32 { uint4 u[2]; v16bf v; };   // 32 bytes: one 16-bit WMMA A/B fragment

__device__ __forceinline__ unsigned short f2bf(float f) {
    unsigned u = __float_as_uint(f);
    u += 0x7fffu + ((u >> 16) & 1u);     // round-to-nearest-even
    return (unsigned short)(u >> 16);
}

// CDNA5 async global->LDS copy (ASYNCcnt-tracked, bypasses VGPRs).
// lds_off = LDS byte offset (low 32 bits of generic pointer; LDS aperture
// truncates to addr[31:0] per ISA 10.2).
__device__ __forceinline__ void async_ld_b128(unsigned lds_off,
                                              const unsigned short* g) {
    asm volatile("global_load_async_to_lds_b128 %0, %1, off"
                 :: "v"(lds_off), "v"(g) : "memory");
}
#define WAIT_ASYNC(n) asm volatile("s_wait_asynccnt " #n ::: "memory")
__device__ __forceinline__ unsigned lds_off32(const void* p) {
    return (unsigned)(size_t)p;
}

// --------------------------- fp32 -> bf16 convert ---------------------------
__global__ void cvt_f32_bf16(const float* __restrict__ x,
                             unsigned short* __restrict__ y, int n8) {
    int i = blockIdx.x * blockDim.x + threadIdx.x;
    if (i >= n8) return;
    const float4* x4 = (const float4*)x;
    float4 a = x4[2 * i], b = x4[2 * i + 1];
    uint4 o;
    o.x = (unsigned)f2bf(a.x) | ((unsigned)f2bf(a.y) << 16);
    o.y = (unsigned)f2bf(a.z) | ((unsigned)f2bf(a.w) << 16);
    o.z = (unsigned)f2bf(b.x) | ((unsigned)f2bf(b.y) << 16);
    o.w = (unsigned)f2bf(b.z) | ((unsigned)f2bf(b.w) << 16);
    ((uint4*)y)[i] = o;
}

// ---------------- fp32 [K][N] -> bf16 [N][K] transpose-convert ---------------
__global__ void transpose_cvt(const float* __restrict__ W,
                              unsigned short* __restrict__ Wt, int K, int N) {
    __shared__ unsigned short tile[32][33];
    int k0 = blockIdx.y * 32, n0 = blockIdx.x * 32;
    int tx = threadIdx.x, ty = threadIdx.y;        // block (32,8)
    #pragma unroll
    for (int j = 0; j < 32; j += 8)
        tile[ty + j][tx] = f2bf(W[(size_t)(k0 + ty + j) * N + n0 + tx]);
    __syncthreads();
    #pragma unroll
    for (int j = 0; j < 32; j += 8)
        Wt[(size_t)(n0 + ty + j) * K + k0 + tx] = tile[tx][ty + j];
}

// ------------------- rel_table row-sum: [4095][64] -> [4095] -----------------
__global__ void relsum_kernel(const float* __restrict__ tab,
                              float* __restrict__ out, int rows) {
    int r = blockIdx.x * blockDim.x + threadIdx.x;
    if (r >= rows) return;
    const float4* t4 = (const float4*)(tab + (size_t)r * 64);
    float s = 0.f;
    #pragma unroll
    for (int i = 0; i < 16; i++) { float4 v = t4[i]; s += v.x + v.y + v.z + v.w; }
    out[r] = s;
}

// --------------------------- bf16 WMMA GEMM ---------------------------------
// C[M][N] = A[M][K] (bf16, row-major) @ Bt[N][K]^T (bf16, N-major) + bias[N]
// 128x128 workgroup tile, 8 waves (4x2), each wave 32x64 (2x4 16x16 tiles).
// K-step 32, double-buffered LDS staged via global_load_async_to_lds_b128.
__global__ __launch_bounds__(256) void gemm_bf16_wmma(
    const unsigned short* __restrict__ A,
    const unsigned short* __restrict__ Bt,
    const float* __restrict__ bias,
    float* __restrict__ Cf, unsigned short* __restrict__ Cb,
    int M, int N, int K, int store_f32) {
    __shared__ __align__(16) unsigned short As[2][128 * 32];
    __shared__ __align__(16) unsigned short Bs[2][128 * 32];
    const int tid = threadIdx.x;
    const int lane = tid & 31, w = tid >> 5;
    const int wm = w & 3, wn = w >> 2;
    const int m0 = blockIdx.y * 128, n0 = blockIdx.x * 128;
    const int lrow = lane & 15, kb = (lane >> 4) * 8;

    // staging addresses for this thread (128 rows x 32 bf16 per operand/stage)
    const int srow = tid >> 1;
    const int sch  = (tid & 1) * 16;
    const unsigned short* gA = A  + (size_t)(m0 + srow) * K + sch;
    const unsigned short* gB = Bt + (size_t)(n0 + srow) * K + sch;
    unsigned la0 = lds_off32(&As[0][srow * 32 + sch]);
    unsigned la1 = lds_off32(&As[1][srow * 32 + sch]);
    unsigned lb0 = lds_off32(&Bs[0][srow * 32 + sch]);
    unsigned lb1 = lds_off32(&Bs[1][srow * 32 + sch]);

    v8f zero = {};
    v8f acc[2][4];
    #pragma unroll
    for (int i = 0; i < 2; i++)
        #pragma unroll
        for (int j = 0; j < 4; j++) acc[i][j] = zero;

    // prologue: stage 0
    async_ld_b128(la0,      gA);
    async_ld_b128(la0 + 16, gA + 8);
    async_ld_b128(lb0,      gB);
    async_ld_b128(lb0 + 16, gB + 8);

    for (int kt = 0; kt < K; kt += 32) {
        const int cur = (kt >> 5) & 1;
        if (kt + 32 < K) {   // issue next stage into the alternate buffer
            unsigned na = cur ? la0 : la1;
            unsigned nb = cur ? lb0 : lb1;
            async_ld_b128(na,      gA + kt + 32);
            async_ld_b128(na + 16, gA + kt + 40);
            async_ld_b128(nb,      gB + kt + 32);
            async_ld_b128(nb + 16, gB + kt + 40);
            WAIT_ASYNC(0x4);   // async ops complete in order: current stage done
        } else {
            WAIT_ASYNC(0x0);
        }
        __syncthreads();       // current stage visible to all waves

        Frag32 af[2], bfr[4];
        #pragma unroll
        for (int mt = 0; mt < 2; mt++) {
            int r = wm * 32 + mt * 16 + lrow;
            af[mt].u[0] = *(const uint4*)&As[cur][r * 32 + kb];
            af[mt].u[1] = *(const uint4*)&As[cur][r * 32 + kb + 16];
        }
        #pragma unroll
        for (int nt = 0; nt < 4; nt++) {
            int r = wn * 64 + nt * 16 + lrow;
            bfr[nt].u[0] = *(const uint4*)&Bs[cur][r * 32 + kb];
            bfr[nt].u[1] = *(const uint4*)&Bs[cur][r * 32 + kb + 16];
        }
        #pragma unroll
        for (int mt = 0; mt < 2; mt++)
            #pragma unroll
            for (int nt = 0; nt < 4; nt++)
                acc[mt][nt] = __builtin_amdgcn_wmma_f32_16x16x32_bf16(
                    false, af[mt].v, false, bfr[nt].v, (short)0, acc[mt][nt],
                    false, false);
        __syncthreads();       // all waves done reading: buffer may be rewritten
    }

    #pragma unroll
    for (int nt = 0; nt < 4; nt++) {
        int col = n0 + wn * 64 + nt * 16 + lrow;
        float bv = bias[col];
        #pragma unroll
        for (int mt = 0; mt < 2; mt++)
            #pragma unroll
            for (int r = 0; r < 8; r++) {
                int row = m0 + wm * 32 + mt * 16 + ((lane & 16) ? r + 8 : r);
                float v = acc[mt][nt][r] + bv;
                if (store_f32) Cf[(size_t)row * N + col] = v;
                else           Cb[(size_t)row * N + col] = f2bf(v);
            }
    }
}

// ------------------------- flash attention (bf16 WMMA) -----------------------
// grid: (S/128, B*H). 8 waves/WG; wave owns 16 query rows. Key blocks of 32.
__global__ __launch_bounds__(256) void attn_kernel(
    const unsigned short* __restrict__ Qp,
    const unsigned short* __restrict__ Kp,
    const unsigned short* __restrict__ Vp,
    const float* __restrict__ relsum,
    unsigned short* __restrict__ Y) {
    __shared__ __align__(16) unsigned short Kt[32 * 64];        // K block row-major
    __shared__ __align__(16) unsigned short Vt[64 * 32];        // V block transposed
    __shared__ __align__(16) unsigned short Pl[8 * 16 * 32];    // per-wave P tiles

    const int tid = threadIdx.x, lane = tid & 31, w = tid >> 5;
    const int bh = blockIdx.y, b = bh >> 4, h = bh & 15;
    const int qwg = blockIdx.x * 128;
    const int qr0 = qwg + w * 16;
    const int lrow = lane & 15, kb = (lane >> 4) * 8;
    const int rsel = (lane & 16) ? 8 : 0;

    // Q fragments for this wave's 16 rows (2 chunks of K=32), kept in registers
    Frag32 qf[2];
    {
        const size_t qoff = (size_t)(b * S_ + qr0 + lrow) * D_ + h * DK_;
        #pragma unroll
        for (int kc = 0; kc < 2; kc++) {
            qf[kc].u[0] = *(const uint4*)(Qp + qoff + kc * 32 + kb);
            qf[kc].u[1] = *(const uint4*)(Qp + qoff + kc * 32 + kb + 16);
        }
    }

    float mrow[8], lsum[8];
    #pragma unroll
    for (int r = 0; r < 8; r++) { mrow[r] = -3.0e38f; lsum[r] = 0.f; }
    v8f zero = {};
    v8f oacc[4];
    #pragma unroll
    for (int nt = 0; nt < 4; nt++) oacc[nt] = zero;

    unsigned short* Pw = &Pl[w * 16 * 32];
    const int kmax = qwg + 128;   // causal: no key block past the wg diagonal

    // per-thread staging addresses (32 rows x 64 bf16 per block)
    const int sk = tid >> 3, sc = (tid & 7) * 8;
    const unsigned ldsK = lds_off32(&Kt[sk * 64 + sc]);

    for (int k0 = 0; k0 < kmax; k0 += 32) {
        const size_t roff = (size_t)(b * S_ + k0 + sk) * D_ + h * DK_ + sc;
        // K tile: direct async global->LDS (row-major == N-major operand of QK^T)
        async_ld_b128(ldsK, Kp + roff);
        // V tile: transpose through registers into [DK][32]
        {
            uint4 vv = *(const uint4*)(Vp + roff);
            const unsigned short* ve = (const unsigned short*)&vv;
            #pragma unroll
            for (int j = 0; j < 8; j++) Vt[(sc + j) * 32 + sk] = ve[j];
        }
        WAIT_ASYNC(0x0);
        __syncthreads();

        // S = Q(16x64) . K^T -> 16x32 in two 16x16 tiles
        v8f sacc[2];
        #pragma unroll
        for (int nt = 0; nt < 2; nt++) {
            sacc[nt] = zero;
            #pragma unroll
            for (int kc = 0; kc < 2; kc++) {
                Frag32 kf;
                int r = nt * 16 + lrow;
                kf.u[0] = *(const uint4*)&Kt[r * 64 + kc * 32 + kb];
                kf.u[1] = *(const uint4*)&Kt[r * 64 + kc * 32 + kb + 16];
                sacc[nt] = __builtin_amdgcn_wmma_f32_16x16x32_bf16(
                    false, qf[kc].v, false, kf.v, (short)0, sacc[nt], false, false);
            }
        }

        // scale + rel bias + causal mask + online softmax (row stats per half-wave)
        #pragma unroll
        for (int r = 0; r < 8; r++) {
            int i  = qr0 + rsel + r;
            int j0 = k0 + lrow;
            int j1 = j0 + 16;
            float s0 = sacc[0][r] * 0.125f + relsum[j0 - i + MAXREL_];
            float s1 = sacc[1][r] * 0.125f + relsum[j1 - i + MAXREL_];
            if (j0 > i) s0 = -1e30f;
            if (j1 > i) s1 = -1e30f;
            float mloc = fmaxf(s0, s1);
            #pragma unroll
            for (int d = 8; d >= 1; d >>= 1)
                mloc = fmaxf(mloc, __shfl_xor(mloc, d, 32));
            float mnew  = fmaxf(mrow[r], mloc);
            float alpha = __expf(mrow[r] - mnew);
            float e0 = __expf(s0 - mnew), e1 = __expf(s1 - mnew);
            float rs = e0 + e1;
            #pragma unroll
            for (int d = 8; d >= 1; d >>= 1) rs += __shfl_xor(rs, d, 32);
            lsum[r] = lsum[r] * alpha + rs;
            mrow[r] = mnew;
            #pragma unroll
            for (int nt = 0; nt < 4; nt++) oacc[nt][r] = oacc[nt][r] * alpha;
            Pw[(rsel + r) * 32 + lrow]      = f2bf(e0);   // C-layout -> A-layout via LDS
            Pw[(rsel + r) * 32 + 16 + lrow] = f2bf(e1);
        }

        // O += P(16x32) . V(32x64); Vt gives the N-major operand directly
        Frag32 pf;
        pf.u[0] = *(const uint4*)&Pw[lrow * 32 + kb];
        pf.u[1] = *(const uint4*)&Pw[lrow * 32 + kb + 16];
        #pragma unroll
        for (int nt = 0; nt < 4; nt++) {
            Frag32 vf;
            int d = nt * 16 + lrow;
            vf.u[0] = *(const uint4*)&Vt[d * 32 + kb];
            vf.u[1] = *(const uint4*)&Vt[d * 32 + kb + 16];
            oacc[nt] = __builtin_amdgcn_wmma_f32_16x16x32_bf16(
                false, pf.v, false, vf.v, (short)0, oacc[nt], false, false);
        }
        __syncthreads();
    }

    // epilogue: O / l -> bf16 into Y (row-major [B*S][D], head-offset columns)
    #pragma unroll
    for (int r = 0; r < 8; r++) {
        float inv = 1.f / lsum[r];
        int row = qr0 + rsel + r;
        #pragma unroll
        for (int nt = 0; nt < 4; nt++) {
            int col = h * DK_ + nt * 16 + lrow;
            Y[(size_t)(b * S_ + row) * D_ + col] = f2bf(oacc[nt][r] * inv);
        }
    }
}

// ------------------------------- launcher -----------------------------------
extern "C" void kernel_launch(void* const* d_in, const int* in_sizes, int n_in,
                              void* d_out, int out_size, void* d_ws, size_t ws_size,
                              hipStream_t stream) {
    const float* q   = (const float*)d_in[0];
    const float* k   = (const float*)d_in[1];
    const float* v   = (const float*)d_in[2];
    const float* Wq  = (const float*)d_in[3];
    const float* bq  = (const float*)d_in[4];
    const float* Wk  = (const float*)d_in[5];
    const float* bk  = (const float*)d_in[6];
    const float* Wv  = (const float*)d_in[7];
    const float* bv  = (const float*)d_in[8];
    const float* Wo  = (const float*)d_in[9];
    const float* bo  = (const float*)d_in[10];
    const float* rel = (const float*)d_in[11];
    // d_in[12] = mask: causal, applied analytically in attn_kernel

    char* ws = (char*)d_ws;
    size_t off = 0;
    auto alloc = [&](size_t bytes) {
        size_t o = off; off += (bytes + 255) & ~(size_t)255; return o;
    };
    const size_t xb = (size_t)B_ * S_ * D_ * sizeof(unsigned short);
    const size_t wb = (size_t)D_ * D_ * sizeof(unsigned short);
    unsigned short* Xq  = (unsigned short*)(ws + alloc(xb));
    unsigned short* Xk  = (unsigned short*)(ws + alloc(xb));
    unsigned short* Xv  = (unsigned short*)(ws + alloc(xb));
    unsigned short* Wtq = (unsigned short*)(ws + alloc(wb));
    unsigned short* Wtk = (unsigned short*)(ws + alloc(wb));
    unsigned short* Wtv = (unsigned short*)(ws + alloc(wb));
    unsigned short* Wto = (unsigned short*)(ws + alloc(wb));
    unsigned short* Qp  = (unsigned short*)(ws + alloc(xb));
    unsigned short* Kp  = (unsigned short*)(ws + alloc(xb));
    unsigned short* Vp  = (unsigned short*)(ws + alloc(xb));
    unsigned short* Yb  = (unsigned short*)(ws + alloc(xb));
    float* rsum = (float*)(ws + alloc((2 * MAXREL_ + 1) * sizeof(float)));

    const int M = B_ * S_, N = D_, K = D_;

    // 1) activation conversion
    int n8 = (B_ * S_ * D_) / 8;
    cvt_f32_bf16<<<n8 / 256, 256, 0, stream>>>(q, Xq, n8);
    cvt_f32_bf16<<<n8 / 256, 256, 0, stream>>>(k, Xk, n8);
    cvt_f32_bf16<<<n8 / 256, 256, 0, stream>>>(v, Xv, n8);

    // 2) weight transpose-convert
    dim3 tg(D_ / 32, D_ / 32), tb(32, 8);
    transpose_cvt<<<tg, tb, 0, stream>>>(Wq, Wtq, D_, D_);
    transpose_cvt<<<tg, tb, 0, stream>>>(Wk, Wtk, D_, D_);
    transpose_cvt<<<tg, tb, 0, stream>>>(Wv, Wtv, D_, D_);
    transpose_cvt<<<tg, tb, 0, stream>>>(Wo, Wto, D_, D_);

    // 3) relative-bias row sums
    relsum_kernel<<<(2 * MAXREL_ + 1 + 255) / 256, 256, 0, stream>>>(rel, rsum, 2 * MAXREL_ + 1);

    // 4) QKV projections (bf16 out)
    dim3 gg(N / 128, M / 128);
    gemm_bf16_wmma<<<gg, 256, 0, stream>>>(Xq, Wtq, bq, nullptr, Qp, M, N, K, 0);
    gemm_bf16_wmma<<<gg, 256, 0, stream>>>(Xk, Wtk, bk, nullptr, Kp, M, N, K, 0);
    gemm_bf16_wmma<<<gg, 256, 0, stream>>>(Xv, Wtv, bv, nullptr, Vp, M, N, K, 0);

    // 5) attention
    dim3 ag(S_ / 128, B_ * H_);
    attn_kernel<<<ag, 256, 0, stream>>>(Qp, Kp, Vp, rsum, Yb);

    // 6) output projection (f32 out to d_out)
    gemm_bf16_wmma<<<gg, 256, 0, stream>>>(Yb, Wto, bo, (float*)d_out, nullptr, M, N, K, 1);
}
